// ST_decomposition_5832565587989
// MI455X (gfx1250) — matrix operand order repaired and checked by reading, not compiled, not run on hardware
//
#include <hip/hip_runtime.h>
#include <hip/hip_bf16.h>

#define B_   4
#define L_   64
#define N_   207
#define C_   64
#define K_   16
#define NH_  4
#define HD_  16
#define NPAD 224
#define TOK  52992      // B*L*N == B*N*L
#define BL_  256
#define BN_  828
#define DD_  13248      // C*N

typedef _Float16 h16;
typedef __attribute__((ext_vector_type(16))) _Float16 v16h;
typedef __attribute__((ext_vector_type(8)))  float    v8f;

// ---------------------------------------------------------------- WMMA core
__device__ __forceinline__ v8f wmma32(v16h a, v16h b, v8f c) {
  return __builtin_amdgcn_wmma_f32_16x16x32_f16(false, a, false, b, (short)0, c, false, false);
}

// A-fragment (16x32 f16, row-major source, ld in halves). A points at (row0,k0).
// Per lane: two contiguous 8-half (16B) segments -> vectorizes to b128 loads.
__device__ __forceinline__ v16h frag_a(const h16* A, int ld, int lane) {
  int m = lane & 15, kb = (lane < 16) ? 0 : 8;
  const h16* p = A + (size_t)m * ld + kb;
  v16h r;
#pragma unroll
  for (int j = 0; j < 8; ++j) { r[j] = p[j]; r[8 + j] = p[16 + j]; }
  return r;
}
// B-fragment from K-contiguous (transposed) storage S[n][k] (ld in halves).
// S points at (n0,k0). Per lane: two contiguous 8-half (16B) segments -> 2x b128 loads.
__device__ __forceinline__ v16h frag_bt(const h16* S, int ld, int lane) {
  int n = lane & 15, kb = (lane < 16) ? 0 : 8;
  const h16* p = S + (size_t)n * ld + kb;
  v16h r;
#pragma unroll
  for (int j = 0; j < 8; ++j) { r[j] = p[j]; r[8 + j] = p[16 + j]; }
  return r;
}
// wave-local LDS ordering (per-wave tile loops are divergent across waves, so no s_barrier)
__device__ __forceinline__ void wave_fence() {
  __builtin_amdgcn_wave_barrier();
  asm volatile("s_wait_dscnt 0x0" ::: "memory");
  __builtin_amdgcn_wave_barrier();
}

__device__ __forceinline__ void block_reduce_add(float v, float* target) {
  __shared__ float red[256];
  red[threadIdx.x] = v; __syncthreads();
  for (int o = 128; o > 0; o >>= 1) {
    if ((int)threadIdx.x < o) red[threadIdx.x] += red[threadIdx.x + o];
    __syncthreads();
  }
  if (threadIdx.x == 0) atomicAdd(target, red[0]);
}

// ---------------------------------------------------------------- prep kernels
__global__ void k_zero(float* p, int n) { int i = blockIdx.x * 256 + threadIdx.x; if (i < n) p[i] = 0.f; }
__global__ void k_f32_to_f16(const float* in, h16* out, int n) {
  int i = blockIdx.x * 256 + threadIdx.x; if (i < n) out[i] = (h16)in[i];
}
// row-major (K,Nn) fp32 weight -> transposed f16 [Nn][K]
__global__ void k_cvt_T(const float* in, h16* out, int K, int Nn) {
  int i = blockIdx.x * 256 + threadIdx.x;
  if (i < K * Nn) { int n = i / K, k = i % K; out[i] = (h16)in[k * Nn + n]; }
}
// sa_W (64x8) -> transposed padded [16][64]
__global__ void k_pad_saT(const float* in, h16* out) {
  int i = blockIdx.x * 256 + threadIdx.x;
  if (i < 16 * 64) { int n = i / 64, k = i % 64; out[i] = (n < 8) ? (h16)in[k * 8 + n] : (h16)0.f; }
}
// conv OIHW(64,64,3,3) -> [tap][cout][cin] f16 (K-contiguous per output row)
__global__ void k_conv_wT(const float* in, h16* out) {
  int idx = blockIdx.x * 256 + threadIdx.x;
  if (idx < 64 * 64 * 9) {
    int tap = idx / 4096, r = idx % 4096, co = r / 64, ci = r % 64;
    int th = tap / 3, tw = tap % 3;
    out[idx] = (h16)in[((co * 64 + ci) * 3 + th) * 3 + tw];
  }
}
__global__ void k_mask(const float* adj, float* m, int n) {
  int i = blockIdx.x * 256 + threadIdx.x; if (i < n) m[i] = (adj[i] > 0.f) ? 1.f : 0.f;
}

// ---------------------------------------------------------------- elementwise stages
__global__ void k_inproj(const float* x, const float* w, const float* b, float* oF, h16* oH) {
  int tok = blockIdx.x, c = threadIdx.x;                 // token order (b,n,l)
  int bb = tok / (N_ * L_), rr = tok % (N_ * L_), n = rr / L_, l = rr % L_;
  float v = w[c] * x[(bb * L_ + l) * N_ + n] + b[c];
  size_t o = (size_t)tok * C_ + c; oF[o] = v; oH[o] = (h16)v;
}
__global__ void k_perm_tf(const float* in, float* oF, h16* oH) { // (b,n,l)->(b,l,n) token order
  int t = blockIdx.x, c = threadIdx.x;
  int bb = t / (L_ * N_), rr = t % (L_ * N_), l = rr / N_, n = rr % N_;
  float v = in[(((size_t)bb * N_ + n) * L_ + l) * C_ + c];
  size_t o = (size_t)t * C_ + c; oF[o] = v; oH[o] = (h16)v;
}
__global__ void k_ln_res(const float* X, const float* Y, const float* g, const float* b,
                         float* oF, h16* oH, int M) {
  int t = blockIdx.x * 256 + threadIdx.x; if (t >= M) return;
  const float* xp = X + (size_t)t * C_; const float* yp = Y + (size_t)t * C_;
  float s = 0.f;
  for (int c = 0; c < C_; ++c) s += xp[c] + yp[c];
  float m = s / C_, v = 0.f;
  for (int c = 0; c < C_; ++c) { float d = xp[c] + yp[c] - m; v += d * d; }
  float inv = rsqrtf(v / C_ + 1e-5f);
  for (int c = 0; c < C_; ++c) {
    float o = (xp[c] + yp[c] - m) * inv * g[c] + b[c];
    oF[(size_t)t * C_ + c] = o; oH[(size_t)t * C_ + c] = (h16)o;
  }
}

// ---------------------------------------------------------------- generic WMMA GEMM
// out = act(A[M,K]f16 @ W + bias); W pre-packed transposed [Nn][K].
// act: 0 none, 1 relu, 2 exact gelu
__global__ void k_gemm(const h16* __restrict__ A, const h16* __restrict__ WT,
                       const float* __restrict__ bias, float* __restrict__ oF,
                       h16* __restrict__ oH, int M, int K, int Nn, int act) {
  int wave = threadIdx.x >> 5, lane = threadIdx.x & 31;
  int row0 = blockIdx.x * 64 + wave * 16;
  if (row0 >= M) return;
  for (int nt = 0; nt * 16 < Nn; ++nt) {
    v8f acc = {};
    for (int k0 = 0; k0 < K; k0 += 32) {
      v16h a = frag_a(A + (size_t)row0 * K + k0, K, lane);
      v16h b = frag_bt(WT + (size_t)(nt * 16) * K + k0, K, lane);
      acc = wmma32(a, b, acc);
    }
    int col = nt * 16 + (lane & 15), rb = row0 + ((lane < 16) ? 0 : 8);
#pragma unroll
    for (int i = 0; i < 8; ++i) {
      float v = acc[i];
      if (bias) v += bias[col];
      if (act == 1) v = fmaxf(v, 0.f);
      else if (act == 2) v = 0.5f * v * (1.f + erff(v * 0.70710678f));
      size_t o = (size_t)(rb + i) * Nn + col;
      if (oF) oF[o] = v;
      if (oH) oH[o] = (h16)v;
    }
  }
}

// ---------------------------------------------------------------- fused temporal attention
__global__ void k_attn_time(const h16* __restrict__ Q, const h16* __restrict__ Km,
                            const h16* __restrict__ V, h16* __restrict__ O) {
  __shared__ h16 sQ[64 * 32]; __shared__ h16 sK[64 * 32]; __shared__ h16 sVt[16 * 64];
  __shared__ float sS[64 * 64]; __shared__ h16 sP[64 * 64];
  int bn = blockIdx.x / NH_, h = blockIdx.x % NH_;
  size_t base = (size_t)bn * L_;
  for (int i = threadIdx.x; i < 64 * 32; i += 128) {
    int r = i >> 5, c = i & 31;
    sQ[i] = (c < 16) ? Q[(base + r) * C_ + h * 16 + c] : (h16)0.f;
    sK[i] = (c < 16) ? Km[(base + r) * C_ + h * 16 + c] : (h16)0.f;
  }
  for (int i = threadIdx.x; i < 16 * 64; i += 128) {   // transposed: [dim][key]
    int c = i >> 6, r = i & 63;
    sVt[i] = V[(base + r) * C_ + h * 16 + c];
  }
  __syncthreads();
  int wave = threadIdx.x >> 5, lane = threadIdx.x & 31;
  int row0 = wave * 16;
  for (int nt = 0; nt < 4; ++nt) {
    v16h a = frag_a(sQ + row0 * 32, 32, lane);
    v16h b = frag_bt(sK + nt * 16 * 32, 32, lane);
    v8f acc = {};
    acc = wmma32(a, b, acc);
    int col = nt * 16 + (lane & 15), rb = row0 + ((lane < 16) ? 0 : 8);
#pragma unroll
    for (int i = 0; i < 8; ++i) sS[(rb + i) * 64 + col] = acc[i] * 0.25f;  // 1/sqrt(HD)
  }
  __syncthreads();
  if (threadIdx.x < 64) {
    float* rp = sS + threadIdx.x * 64;
    float mx = rp[0];
    for (int i = 1; i < 64; ++i) mx = fmaxf(mx, rp[i]);
    float s = 0.f;
    for (int i = 0; i < 64; ++i) { float e = __expf(rp[i] - mx); rp[i] = e; s += e; }
    float inv = 1.f / s;
    for (int i = 0; i < 64; ++i) sP[threadIdx.x * 64 + i] = (h16)(rp[i] * inv);
  }
  __syncthreads();
  v8f acc = {};
  for (int k0 = 0; k0 < 64; k0 += 32) {
    v16h a = frag_a(sP + row0 * 64 + k0, 64, lane);
    v16h b = frag_bt(sVt + k0, 64, lane);
    acc = wmma32(a, b, acc);
  }
  int col = lane & 15, rb = row0 + ((lane < 16) ? 0 : 8);
#pragma unroll
  for (int i = 0; i < 8; ++i) O[(base + rb + i) * C_ + h * 16 + col] = (h16)acc[i];
}

// Per-wave softmax over a 16xNPAD f32 score tile; packs f16 probs into the first
// half of each row's own f32 storage (write@2i trails read@4i within one lane).
// Resulting prob matrix: halves, row stride 2*NPAD (=448), K contiguous.
__device__ __forceinline__ void wave_softmax_pack(float* mS, int lane) {
  if (lane < 16) {
    float* rp = mS + lane * NPAD;
    h16* ph = (h16*)rp;
    float mx = -3.4e38f;
    for (int i = 0; i < NPAD; ++i) mx = fmaxf(mx, rp[i]);
    float s = 0.f;
    for (int i = 0; i < NPAD; ++i) { float e = __expf(rp[i] - mx); rp[i] = e; s += e; }
    float inv = 1.f / s;
    for (int i = 0; i < NPAD; ++i) ph[i] = (h16)(rp[i] * inv);
  }
}

// ---------------------------------------------------------------- masked feature attention
__global__ void k_attn_feat(const h16* __restrict__ Q, const h16* __restrict__ Km,
                            const h16* __restrict__ V, const float* __restrict__ mask,
                            h16* __restrict__ O) {
  __shared__ h16 sK[NPAD * 32]; __shared__ h16 sVt[16 * NPAD];
  __shared__ float sS[4 * 16 * NPAD];
  int bl = blockIdx.x / NH_, h = blockIdx.x % NH_;
  size_t base = (size_t)bl * N_;
  for (int i = threadIdx.x; i < NPAD * 32; i += 128) {
    int r = i >> 5, c = i & 31;
    sK[i] = (r < N_ && c < 16) ? Km[(base + r) * C_ + h * 16 + c] : (h16)0.f;
  }
  for (int i = threadIdx.x; i < 16 * NPAD; i += 128) {  // transposed: [dim][key]
    int c = i / NPAD, r = i % NPAD;
    sVt[i] = (r < N_) ? V[(base + r) * C_ + h * 16 + c] : (h16)0.f;
  }
  __syncthreads();
  int wave = threadIdx.x >> 5, lane = threadIdx.x & 31;
  float* mS = sS + wave * 16 * NPAD;
  for (int mt = wave; mt < 13; mt += 4) {
    int row0 = mt * 16;
    int qr = row0 + (lane & 15), kb = (lane < 16) ? 0 : 8;
    v16h aq;
    if (qr < N_) {
      const h16* p = Q + (base + qr) * C_ + h * 16 + kb;
#pragma unroll
      for (int j = 0; j < 8; ++j) { aq[j] = p[j]; aq[8 + j] = (h16)0.f; }
    } else {
#pragma unroll
      for (int j = 0; j < 16; ++j) aq[j] = (h16)0.f;
    }
    for (int nt = 0; nt < 14; ++nt) {
      v16h b = frag_bt(sK + nt * 16 * 32, 32, lane);
      v8f acc = {};
      acc = wmma32(aq, b, acc);
      int gc = nt * 16 + (lane & 15), rb = (lane < 16) ? 0 : 8;
#pragma unroll
      for (int i = 0; i < 8; ++i) {
        int gr = row0 + rb + i;
        float v = acc[i] * 0.25f;
        if (gr >= N_ || gc >= N_ || mask[gr * N_ + gc] == 0.f) v = -1e9f;
        mS[(rb + i) * NPAD + gc] = v;
      }
    }
    wave_fence();
    wave_softmax_pack(mS, lane);
    wave_fence();
    v8f acc = {};
    for (int k0 = 0; k0 < NPAD; k0 += 32) {
      v16h a = frag_a((const h16*)mS + k0, 2 * NPAD, lane);
      v16h b = frag_bt(sVt + k0, NPAD, lane);
      acc = wmma32(a, b, acc);
    }
    int col = lane & 15, rb = (lane < 16) ? 0 : 8;
#pragma unroll
    for (int i = 0; i < 8; ++i) {
      int gr = row0 + rb + i;
      if (gr < N_) O[(base + gr) * C_ + h * 16 + col] = (h16)acc[i];
    }
    wave_fence();
  }
}

// ---------------------------------------------------------------- SAGAN spatial attention
__global__ void k_attn_spatial(const h16* __restrict__ Qs, const h16* __restrict__ Ks,
                               const h16* __restrict__ Vs, const float* __restrict__ Yt,
                               const float* __restrict__ gamma,
                               float* __restrict__ ZF, h16* __restrict__ ZH) {
  __shared__ h16 sK[NPAD * 32]; __shared__ h16 sVt[64 * NPAD];
  __shared__ float sS[4 * 16 * NPAD];
  size_t base = (size_t)blockIdx.x * N_;
  for (int i = threadIdx.x; i < NPAD * 32; i += 128) {
    int r = i >> 5, c = i & 31;
    sK[i] = (r < N_ && c < 16) ? Ks[(base + r) * 16 + c] : (h16)0.f;
  }
  for (int i = threadIdx.x; i < 64 * NPAD; i += 128) {  // transposed: [dim][key]
    int c = i / NPAD, r = i % NPAD;
    sVt[i] = (r < N_) ? Vs[(base + r) * 64 + c] : (h16)0.f;
  }
  __syncthreads();
  int wave = threadIdx.x >> 5, lane = threadIdx.x & 31;
  float* mS = sS + wave * 16 * NPAD;
  float g = gamma[0];
  for (int mt = wave; mt < 13; mt += 4) {
    int row0 = mt * 16;
    int qr = row0 + (lane & 15), kb = (lane < 16) ? 0 : 8;
    v16h aq;
    if (qr < N_) {
      const h16* p = Qs + (base + qr) * 16 + kb;
#pragma unroll
      for (int j = 0; j < 8; ++j) { aq[j] = p[j]; aq[8 + j] = (h16)0.f; }
    } else {
#pragma unroll
      for (int j = 0; j < 16; ++j) aq[j] = (h16)0.f;
    }
    for (int nt = 0; nt < 14; ++nt) {
      v16h b = frag_bt(sK + nt * 16 * 32, 32, lane);
      v8f acc = {};
      acc = wmma32(aq, b, acc);
      int gc = nt * 16 + (lane & 15), rb = (lane < 16) ? 0 : 8;
#pragma unroll
      for (int i = 0; i < 8; ++i) {
        int gr = row0 + rb + i;
        float v = acc[i];
        if (gr >= N_ || gc >= N_) v = -1e9f;
        mS[(rb + i) * NPAD + gc] = v;
      }
    }
    wave_fence();
    wave_softmax_pack(mS, lane);
    wave_fence();
    for (int nt = 0; nt < 4; ++nt) {
      v8f acc = {};
      for (int k0 = 0; k0 < NPAD; k0 += 32) {
        v16h a = frag_a((const h16*)mS + k0, 2 * NPAD, lane);
        v16h b = frag_bt(sVt + (size_t)(nt * 16) * NPAD + k0, NPAD, lane);
        acc = wmma32(a, b, acc);
      }
      int col = nt * 16 + (lane & 15), rb = (lane < 16) ? 0 : 8;
#pragma unroll
      for (int i = 0; i < 8; ++i) {
        int gr = row0 + rb + i;
        if (gr < N_) {
          size_t o = (base + gr) * (size_t)64 + col;
          float v = g * acc[i] + Yt[o];
          ZF[o] = v; ZH[o] = (h16)v;
        }
      }
    }
    wave_fence();
  }
}

// ---------------------------------------------------------------- GAT
__global__ void k_gat_e(const h16* Wh, const float* a, float* e1, float* e2, int M) {
  int t = blockIdx.x * 256 + threadIdx.x; if (t >= M) return;
  float s1 = 0.f, s2 = 0.f;
  for (int c = 0; c < C_; ++c) {
    float w = (float)Wh[(size_t)t * C_ + c];
    s1 += w * a[c]; s2 += w * a[64 + c];
  }
  e1[t] = s1; e2[t] = s2;
}

__global__ void k_gat_agg(const float* __restrict__ e1, const float* __restrict__ e2,
                          const h16* __restrict__ Wh, const float* __restrict__ mask,
                          h16* __restrict__ Hout) {
  __shared__ h16 sWt[64 * NPAD];
  __shared__ float sS[4 * 16 * NPAD];
  size_t base = (size_t)blockIdx.x * N_;
  for (int i = threadIdx.x; i < 64 * NPAD; i += 128) {  // transposed: [dim][node]
    int c = i / NPAD, r = i % NPAD;
    sWt[i] = (r < N_) ? Wh[(base + r) * 64 + c] : (h16)0.f;
  }
  __syncthreads();
  int wave = threadIdx.x >> 5, lane = threadIdx.x & 31;
  float* mS = sS + wave * 16 * NPAD;
  for (int mt = wave; mt < 13; mt += 4) {
    int row0 = mt * 16;
    for (int idx = lane; idx < 16 * NPAD; idx += 32) {
      int lr = idx / NPAD, gc = idx % NPAD, gr = row0 + lr;
      float v = -9e15f;
      if (gr < N_ && gc < N_ && mask[gr * N_ + gc] != 0.f) {
        float s = e1[base + gr] + e2[base + gc];
        v = (s > 0.f) ? s : 0.2f * s;                  // leaky_relu(0.2)
      }
      mS[lr * NPAD + gc] = v;
    }
    wave_fence();
    wave_softmax_pack(mS, lane);
    wave_fence();
    for (int nt = 0; nt < 4; ++nt) {
      v8f acc = {};
      for (int k0 = 0; k0 < NPAD; k0 += 32) {
        v16h a = frag_a((const h16*)mS + k0, 2 * NPAD, lane);
        v16h b = frag_bt(sWt + (size_t)(nt * 16) * NPAD + k0, NPAD, lane);
        acc = wmma32(a, b, acc);
      }
      int col = nt * 16 + (lane & 15), rb = (lane < 16) ? 0 : 8;
#pragma unroll
      for (int i = 0; i < 8; ++i) {
        int gr = row0 + rb + i;
        if (gr < N_) {
          float v = acc[i];
          v = (v > 0.f) ? v : (__expf(v) - 1.f);       // elu
          Hout[(base + gr) * 64 + col] = (h16)v;
        }
      }
    }
    wave_fence();
  }
}

// ---------------------------------------------------------------- VQ
__global__ void k_vq_assign(const float* zt, const float* rflat, int mode,
                            const float* emb, int* idx) {
  __shared__ float se[16 * 256]; __shared__ float se2[16];
  for (int i = threadIdx.x; i < 16 * 256; i += 64) se[i] = emb[i];
  __syncthreads();
  if (threadIdx.x < 16) {
    float s = 0.f;
    for (int j = 0; j < 256; ++j) { float e = se[threadIdx.x * 256 + j]; s += e * e; }
    se2[threadIdx.x] = s;
  }
  __syncthreads();
  int d = blockIdx.x * 64 + threadIdx.x; if (d >= DD_) return;
  int c = d / N_, n = d % N_;
  float dot[16];
#pragma unroll
  for (int k = 0; k < 16; ++k) dot[k] = 0.f;
  float z2 = 0.f;
  for (int j = 0; j < 256; ++j) {
    float v = mode ? rflat[(size_t)d * 256 + j] : zt[((size_t)j * N_ + n) * C_ + c];
    z2 += v * v;
#pragma unroll
    for (int k = 0; k < 16; ++k) dot[k] += v * se[k * 256 + j];
  }
  int best = 0; float bd = z2 - 2.f * dot[0] + se2[0];
  for (int k = 1; k < 16; ++k) {
    float dd = z2 - 2.f * dot[k] + se2[k];
    if (dd < bd) { bd = dd; best = k; }
  }
  idx[d] = best;
}

__global__ void k_vq_apply1(const float* zt, const float* emb1, const int* idx1,
                            float* rflat, float* acc) {
  int lin = blockIdx.x * 256 + threadIdx.x;
  float loc = 0.f;
  if (lin < DD_ * 256) {
    int d = lin >> 8, j = lin & 255, c = d / N_, n = d % N_;
    float z = zt[((size_t)j * N_ + n) * C_ + c];
    float q = emb1[idx1[d] * 256 + j];
    float r = z - q; rflat[lin] = r; loc = r * r;
  }
  block_reduce_add(loc, acc);
}

__global__ void k_vq_apply2(const float* zt, const float* rflat, const float* emb2,
                            const int* idx2, h16* qtok, float* acc) {
  int lin = blockIdx.x * 256 + threadIdx.x;
  float loc = 0.f;
  if (lin < DD_ * 256) {
    int d = lin >> 8, j = lin & 255, c = d / N_, n = d % N_;
    float z = zt[((size_t)j * N_ + n) * C_ + c];
    float r = rflat[lin];
    float q2 = emb2[idx2[d] * 256 + j];
    float qsum = (z - r) + q2;
    int bb = j / L_, l = j % L_;
    qtok[(((size_t)(bb * N_ + n) * L_ + l)) * C_ + c] = (h16)qsum;
    float dq = q2 - r; loc = dq * dq;
  }
  block_reduce_add(loc, acc);
}

__global__ void k_hist(const int* idx, float* hist) {
  int d = blockIdx.x * 256 + threadIdx.x;
  if (d < DD_) atomicAdd(&hist[idx[d]], 1.f);
}

// ---------------------------------------------------------------- classifier / residual
__global__ void k_softmax16(float* X, int M) {
  int t = blockIdx.x * 256 + threadIdx.x; if (t >= M) return;
  float* p = X + (size_t)t * 16;
  float mx = p[0];
  for (int k = 1; k < 16; ++k) mx = fmaxf(mx, p[k]);
  float s = 0.f;
  for (int k = 0; k < 16; ++k) { float e = __expf(p[k] - mx); p[k] = e; s += e; }
  float inv = 1.f / s;
  for (int k = 0; k < 16; ++k) p[k] *= inv;
}

__global__ void k_wres(const float* zt, const float* soft, const float* emb2,
                       float* wres_out, h16* restok, float* acc) {
  int t = blockIdx.x * 256 + threadIdx.x;
  float loc = 0.f;
  if (t < TOK) {
    int j = t / N_, n = t % N_, bb = j / L_, l = j % L_;
    float s = 0.f;
    for (int k = 0; k < 16; ++k) s += soft[(size_t)t * 16 + k] * emb2[k * 256 + j];
    size_t dtok = ((size_t)(bb * N_ + n) * L_ + l) * C_;
    for (int c = 0; c < C_; ++c) {
      float w = zt[(size_t)t * C_ + c] - s;
      wres_out[(size_t)t * C_ + c] = w;
      restok[dtok + c] = (h16)w;
      loc += fabsf(w);
    }
  }
  block_reduce_add(loc, acc);
}

// ---------------------------------------------------------------- decoder
__global__ void k_bn_stats(const float* X, float* stats, int M) {
  int ch = blockIdx.x;
  __shared__ float ss[256]; __shared__ float sq[256];
  float s = 0.f, q = 0.f;
  for (int r = threadIdx.x; r < M; r += 256) {
    float v = X[(size_t)r * 64 + ch]; s += v; q += v * v;
  }
  ss[threadIdx.x] = s; sq[threadIdx.x] = q; __syncthreads();
  for (int o = 128; o > 0; o >>= 1) {
    if ((int)threadIdx.x < o) { ss[threadIdx.x] += ss[threadIdx.x + o]; sq[threadIdx.x] += sq[threadIdx.x + o]; }
    __syncthreads();
  }
  if (threadIdx.x == 0) {
    float m = ss[0] / M, var = sq[0] / M - m * m;
    stats[ch * 2] = m; stats[ch * 2 + 1] = rsqrtf(var + 1e-5f);
  }
}

__global__ void k_bn_apply(const float* X, const float* stats, const float* g, const float* be,
                           float* oF, h16* oH, int M) {
  int i = blockIdx.x * 256 + threadIdx.x; if (i >= M * 64) return;
  int ch = i & 63;
  float v = (X[i] - stats[ch * 2]) * stats[ch * 2 + 1] * g[ch] + be[ch];
  v = fmaxf(v, 0.f);
  if (oF) oF[i] = v;
  if (oH) oH[i] = (h16)v;
}

__global__ void k_conv3(const h16* __restrict__ Ain, const h16* __restrict__ Wr,
                        const float* __restrict__ bias, float* __restrict__ oF) {
  int wave = threadIdx.x >> 5, lane = threadIdx.x & 31;
  int row0 = blockIdx.x * 64 + wave * 16;
  int m = lane & 15, kb = (lane < 16) ? 0 : 8;
  int r = row0 + m;
  int bb = r / (N_ * L_), rr = r % (N_ * L_), n = rr / L_, l = rr % L_;
  for (int nt = 0; nt < 4; ++nt) {
    v8f acc = {};
    for (int tap = 0; tap < 9; ++tap) {
      int dn = tap / 3 - 1, dl = tap % 3 - 1;
      int n2 = n + dn, l2 = l + dl;
      bool ok = (n2 >= 0) && (n2 < N_) && (l2 >= 0) && (l2 < L_);
      for (int k0 = 0; k0 < 64; k0 += 32) {
        v16h a;
        if (ok) {
          const h16* p = Ain + (((size_t)(bb * N_ + n2) * L_ + l2) * C_) + k0 + kb;
#pragma unroll
          for (int j = 0; j < 8; ++j) { a[j] = p[j]; a[8 + j] = p[16 + j]; }
        } else {
#pragma unroll
          for (int j = 0; j < 16; ++j) a[j] = (h16)0.f;
        }
        // Wr: [tap][cout][cin], K-contiguous per output row
        v16h bf = frag_bt(Wr + ((size_t)tap * 64 + nt * 16) * 64 + k0, 64, lane);
        acc = wmma32(a, bf, acc);
      }
    }
    int col = nt * 16 + (lane & 15), rb = row0 + ((lane < 16) ? 0 : 8);
#pragma unroll
    for (int i = 0; i < 8; ++i) oF[(size_t)(rb + i) * 64 + col] = acc[i] + bias[col];
  }
}

// ---------------------------------------------------------------- output / losses
__global__ void k_combine(const float* dq, const float* dr, const float* ow, const float* ob,
                          const float* x, float* acc) {
  int t = blockIdx.x * 256 + threadIdx.x;
  float loc = 0.f;
  if (t < TOK) {
    int bb = t / (N_ * L_), rr = t % (N_ * L_), n = rr / L_, l = rr % L_;
    float s = 0.f;
    for (int c = 0; c < C_; ++c) s += (dq[(size_t)t * 64 + c] + dr[(size_t)t * 64 + c]) * ow[c];
    s += ob[0];
    float d = s - x[(bb * L_ + l) * N_ + n];
    loc = d * d;
  }
  block_reduce_add(loc, acc);
}

__global__ void k_dists(const float* fout, const float* emb2, float* out) {
  __shared__ float se[16 * 256]; __shared__ float se2[16];
  for (int i = threadIdx.x; i < 16 * 256; i += 64) se[i] = emb2[i];
  __syncthreads();
  if (threadIdx.x < 16) {
    float s = 0.f;
    for (int j = 0; j < 256; ++j) { float e = se[threadIdx.x * 256 + j]; s += e * e; }
    se2[threadIdx.x] = s;
  }
  __syncthreads();
  int d = blockIdx.x * 64 + threadIdx.x; if (d >= DD_) return;
  int c = d / N_, n = d % N_;
  float dot[16];
#pragma unroll
  for (int k = 0; k < 16; ++k) dot[k] = 0.f;
  float z2 = 0.f;
  for (int j = 0; j < 256; ++j) {
    float v = fout[((size_t)j * N_ + n) * C_ + c];
    z2 += v * v;
#pragma unroll
    for (int k = 0; k < 16; ++k) dot[k] += v * se[k * 256 + j];
  }
  for (int k = 0; k < 16; ++k)
    out[(size_t)d * 16 + k] = sqrtf(fmaxf(z2 - 2.f * dot[k] + se2[k], 0.f));
}

__global__ void k_final(const float* acc, float* out) {
  if (threadIdx.x != 0 || blockIdx.x != 0) return;
  const float NE = (float)DD_ * 256.f;
  float mse1 = acc[0] / NE, mse2 = acc[1] / NE;
  float quant = 1.25f * mse1 + 1.25f * mse2;          // q_loss + COMMIT*e_loss (equal in value)
  float spars = acc[2] / NE;
  float recon = acc[3] / (float)TOK;
  float s1 = 0.f, s2 = 0.f;
  for (int k = 0; k < 16; ++k) {
    float p1 = acc[8 + k] / (float)DD_;  s1 -= p1 * __logf(p1 + 1e-10f);
    float p2 = acc[24 + k] / (float)DD_; s2 -= p2 * __logf(p2 + 1e-10f);
  }
  float pen = -(__expf(s1) + __expf(s2));
  out[0] = recon + 0.5f * quant + 0.05f * spars + 0.01f * pen;
  out[1] = recon; out[2] = quant; out[3] = spars; out[4] = pen;
}

// ---------------------------------------------------------------- host driver
struct TW {
  const h16 *Wq, *Wk, *Wv, *Wo, *Wf1, *Wf2;
  const float *bq, *bk, *bv, *bo, *g1, *b1, *g2, *b2, *bf1, *bf2;
};

static void run_transformer(const TW& w, float* xF, h16* xH, h16* qh, h16* kh, h16* vh,
                            h16* oh, float* tmpF, float* x1F, h16* x1H,
                            float* outF, h16* outH, const float* maskf, bool feat,
                            hipStream_t s) {
  dim3 gB(TOK / 64), tB(128);
  k_gemm<<<gB, tB, 0, s>>>(xH, w.Wq, w.bq, nullptr, qh, TOK, 64, 64, 0);
  k_gemm<<<gB, tB, 0, s>>>(xH, w.Wk, w.bk, nullptr, kh, TOK, 64, 64, 0);
  k_gemm<<<gB, tB, 0, s>>>(xH, w.Wv, w.bv, nullptr, vh, TOK, 64, 64, 0);
  if (!feat) k_attn_time<<<dim3(BN_ * NH_), 128, 0, s>>>(qh, kh, vh, oh);
  else       k_attn_feat<<<dim3(BL_ * NH_), 128, 0, s>>>(qh, kh, vh, maskf, oh);
  k_gemm<<<gB, tB, 0, s>>>(oh, w.Wo, w.bo, tmpF, nullptr, TOK, 64, 64, 0);
  k_ln_res<<<dim3((TOK + 255) / 256), 256, 0, s>>>(xF, tmpF, w.g1, w.b1, x1F, x1H, TOK);
  k_gemm<<<gB, tB, 0, s>>>(x1H, w.Wf1, w.bf1, nullptr, oh, TOK, 64, 64, 2);
  k_gemm<<<gB, tB, 0, s>>>(oh, w.Wf2, w.bf2, tmpF, nullptr, TOK, 64, 64, 0);
  k_ln_res<<<dim3((TOK + 255) / 256), 256, 0, s>>>(x1F, tmpF, w.g2, w.b2, outF, outH, TOK);
}

extern "C" void kernel_launch(void* const* d_in, const int* in_sizes, int n_in,
                              void* d_out, int out_size, void* d_ws, size_t ws_size,
                              hipStream_t stream) {
  (void)in_sizes; (void)n_in; (void)out_size; (void)ws_size;
  enum { I_X = 0, I_ADJ = 1, I_INW = 2, I_INB = 3, I_TIME = 4, I_FEAT = 20,
         I_SAQ = 36, I_SAK = 37, I_SAV = 38, I_SAG = 39,
         I_G0W = 40, I_G0A = 41, I_G1W = 42, I_G1A = 43,
         I_CW1 = 44, I_CB1 = 45, I_CW2 = 46, I_CB2 = 47,
         I_EMB1 = 48, I_EMB2 = 49,
         I_DW1 = 50, I_DB1 = 51, I_DG1 = 52, I_DBE1 = 53,
         I_DW2 = 54, I_DB2 = 55, I_DG2 = 56, I_DBE2 = 57,
         I_RW = 58, I_RB = 59, I_RG = 60, I_RBE = 61, I_OW = 62, I_OB = 63 };
  auto F = [&](int i) { return (const float*)d_in[i]; };
  char* wsp = (char*)d_ws; size_t off = 0;
  auto alloc = [&](size_t bytes) -> char* {
    char* p = wsp + off; off = (off + bytes + 255) & ~(size_t)255; return p;
  };
  const size_t TC = (size_t)TOK * C_;
  float* bufA = (float*)alloc(TC * 4);  float* bufB = (float*)alloc(TC * 4);
  float* bufC = (float*)alloc(TC * 4);  float* bufD = (float*)alloc(TC * 4);
  float* bufE = (float*)alloc(TC * 4);
  h16* hA = (h16*)alloc(TC * 2);  h16* hQ = (h16*)alloc(TC * 2);
  h16* hK = (h16*)alloc(TC * 2);  h16* hV = (h16*)alloc(TC * 2);
  h16* hO = (h16*)alloc(TC * 2);  h16* hX1 = (h16*)alloc(TC * 2);
  h16* hD = (h16*)alloc(TC * 2);  h16* hE = (h16*)alloc(TC * 2);
  h16* hW = (h16*)alloc(TC * 2);  h16* hH = (h16*)alloc(TC * 2);
  h16* qsh = (h16*)alloc((size_t)TOK * 16 * 2);
  h16* ksh = (h16*)alloc((size_t)TOK * 16 * 2);
  float* soft = (float*)alloc((size_t)TOK * 16 * 4);
  float* rflat = (float*)alloc((size_t)DD_ * 256 * 4);
  int* idx1 = (int*)alloc(DD_ * 4);  int* idx2 = (int*)alloc(DD_ * 4);
  float* acc = (float*)alloc(64 * 4);               // [0]e1 [1]e2 [2]|wr| [3]recon, 8..23 hist1, 24..39 hist2
  float* stats = (float*)alloc(3 * 128 * 4);
  float* maskf = (float*)alloc(N_ * N_ * 4);
  h16* wh = (h16*)alloc(120000 * 2);

  // weight f16 arena offsets (all transposed [N][K] layouts)
  const size_t O_TWQ = 0, O_TWK = 4096, O_TWV = 8192, O_TWO = 12288, O_TF1 = 16384, O_TF2 = 20480;
  const size_t O_FWQ = 24576, O_FWK = 28672, O_FWV = 32768, O_FWO = 36864, O_FF1 = 40960, O_FF2 = 45056;
  const size_t O_SAQ = 49152, O_SAK = 50176, O_SAV = 51200;
  const size_t O_G0 = 55296, O_G1 = 59392, O_CW1 = 63488, O_CW2 = 67584;
  const size_t O_DW1T = 68608, O_RWT = 72704, O_W2R = 76800;

  float* out = (float*)d_out;
  float* out_wres = out;
  float* out_centers = out + 3391488;
  float* out_dists = out + 3391488 + 4096;
  float* out_scal = out + 3391488 + 4096 + 211968;

  k_zero<<<1, 64, 0, stream>>>(acc, 64);
  auto cvtT = [&](int idx, size_t o, int K, int Nn) {
    k_cvt_T<<<(K * Nn + 255) / 256, 256, 0, stream>>>(F(idx), wh + o, K, Nn);
  };
  cvtT(I_TIME + 0, O_TWQ, 64, 64); cvtT(I_TIME + 2, O_TWK, 64, 64); cvtT(I_TIME + 4, O_TWV, 64, 64);
  cvtT(I_TIME + 6, O_TWO, 64, 64); cvtT(I_TIME + 12, O_TF1, 64, 64); cvtT(I_TIME + 14, O_TF2, 64, 64);
  cvtT(I_FEAT + 0, O_FWQ, 64, 64); cvtT(I_FEAT + 2, O_FWK, 64, 64); cvtT(I_FEAT + 4, O_FWV, 64, 64);
  cvtT(I_FEAT + 6, O_FWO, 64, 64); cvtT(I_FEAT + 12, O_FF1, 64, 64); cvtT(I_FEAT + 14, O_FF2, 64, 64);
  cvtT(I_SAV, O_SAV, 64, 64);
  cvtT(I_G0W, O_G0, 64, 64); cvtT(I_G1W, O_G1, 64, 64);
  cvtT(I_CW1, O_CW1, 64, 64); cvtT(I_CW2, O_CW2, 64, 16);
  k_pad_saT<<<4, 256, 0, stream>>>(F(I_SAQ), wh + O_SAQ);
  k_pad_saT<<<4, 256, 0, stream>>>(F(I_SAK), wh + O_SAK);
  // dec_w1/res_w are (o,i): out = in @ W^T, so row-major (o,i) IS the transposed [N][K] layout
  k_f32_to_f16<<<16, 256, 0, stream>>>(F(I_DW1), wh + O_DW1T, 4096);
  k_f32_to_f16<<<16, 256, 0, stream>>>(F(I_RW), wh + O_RWT, 4096);
  k_conv_wT<<<144, 256, 0, stream>>>(F(I_DW2), wh + O_W2R);
  k_mask<<<(N_ * N_ + 255) / 256, 256, 0, stream>>>(F(I_ADJ), maskf, N_ * N_);

  // input projection -> temporal token order (b,n,l)
  k_inproj<<<TOK, 64, 0, stream>>>(F(I_X), F(I_INW), F(I_INB), bufA, hA);

  TW tw = { wh + O_TWQ, wh + O_TWK, wh + O_TWV, wh + O_TWO, wh + O_TF1, wh + O_TF2,
            F(I_TIME + 1), F(I_TIME + 3), F(I_TIME + 5), F(I_TIME + 7),
            F(I_TIME + 8), F(I_TIME + 9), F(I_TIME + 10), F(I_TIME + 11),
            F(I_TIME + 13), F(I_TIME + 15) };
  run_transformer(tw, bufA, hA, hQ, hK, hV, hO, bufB, bufC, hX1, bufD, hD, nullptr, false, stream);

  // (b,n,l) -> (b,l,n) token order
  k_perm_tf<<<TOK, 64, 0, stream>>>(bufD, bufA, hA);

  TW fw = { wh + O_FWQ, wh + O_FWK, wh + O_FWV, wh + O_FWO, wh + O_FF1, wh + O_FF2,
            F(I_FEAT + 1), F(I_FEAT + 3), F(I_FEAT + 5), F(I_FEAT + 7),
            F(I_FEAT + 8), F(I_FEAT + 9), F(I_FEAT + 10), F(I_FEAT + 11),
            F(I_FEAT + 13), F(I_FEAT + 15) };
  run_transformer(fw, bufA, hA, hQ, hK, hV, hO, bufB, bufC, hX1, bufD, hD, maskf, true, stream);
  // bufD/hD = s_feat tokens (kept for dists)

  // spatial self-attention
  dim3 gB(TOK / 64), tB(128);
  k_gemm<<<gB, tB, 0, stream>>>(hD, wh + O_SAQ, nullptr, nullptr, qsh, TOK, 64, 16, 0);
  k_gemm<<<gB, tB, 0, stream>>>(hD, wh + O_SAK, nullptr, nullptr, ksh, TOK, 64, 16, 0);
  k_gemm<<<gB, tB, 0, stream>>>(hD, wh + O_SAV, nullptr, nullptr, hV, TOK, 64, 64, 0);
  k_attn_spatial<<<BL_, 128, 0, stream>>>(qsh, ksh, hV, bufD, F(I_SAG), bufE, hE);

  // GAT x2
  k_gemm<<<gB, tB, 0, stream>>>(hE, wh + O_G0, nullptr, nullptr, hW, TOK, 64, 64, 0);
  k_gat_e<<<(TOK + 255) / 256, 256, 0, stream>>>(hW, F(I_G0A), bufB, bufC, TOK);
  k_gat_agg<<<BL_, 128, 0, stream>>>(bufB, bufC, hW, maskf, hH);
  k_gemm<<<gB, tB, 0, stream>>>(hH, wh + O_G1, nullptr, nullptr, hW, TOK, 64, 64, 0);
  k_gat_e<<<(TOK + 255) / 256, 256, 0, stream>>>(hW, F(I_G1A), bufB, bufC, TOK);
  k_gat_agg<<<BL_, 128, 0, stream>>>(bufB, bufC, hW, maskf, hH);

  // classifier soft assignments
  k_gemm<<<gB, tB, 0, stream>>>(hH, wh + O_CW1, F(I_CB1), nullptr, hO, TOK, 64, 64, 1);
  k_gemm<<<gB, tB, 0, stream>>>(hO, wh + O_CW2, F(I_CB2), soft, nullptr, TOK, 64, 16, 0);
  k_softmax16<<<(TOK + 255) / 256, 256, 0, stream>>>(soft, TOK);

  // dual residual VQ
  k_vq_assign<<<207, 64, 0, stream>>>(bufE, rflat, 0, F(I_EMB1), idx1);
  k_vq_apply1<<<(DD_ * 256) / 256, 256, 0, stream>>>(bufE, F(I_EMB1), idx1, rflat, acc + 0);
  k_hist<<<(DD_ + 255) / 256, 256, 0, stream>>>(idx1, acc + 8);
  k_vq_assign<<<207, 64, 0, stream>>>(bufE, rflat, 1, F(I_EMB2), idx2);
  k_vq_apply2<<<(DD_ * 256) / 256, 256, 0, stream>>>(bufE, rflat, F(I_EMB2), idx2, hQ, acc + 1);
  k_hist<<<(DD_ + 255) / 256, 256, 0, stream>>>(idx2, acc + 24);

  // weighted residual (output #1) + residual-branch tokens
  k_wres<<<(TOK + 255) / 256, 256, 0, stream>>>(bufE, soft, F(I_EMB2), out_wres, hK, acc + 2);

  // decoder (quantized path)
  k_gemm<<<gB, tB, 0, stream>>>(hQ, wh + O_DW1T, F(I_DB1), bufA, nullptr, TOK, 64, 64, 0);
  k_bn_stats<<<64, 256, 0, stream>>>(bufA, stats, TOK);
  k_bn_apply<<<(TOK * 64 + 255) / 256, 256, 0, stream>>>(bufA, stats, F(I_DG1), F(I_DBE1),
                                                         nullptr, hV, TOK);
  k_conv3<<<gB, tB, 0, stream>>>(hV, wh + O_W2R, F(I_DB2), bufB);
  k_bn_stats<<<64, 256, 0, stream>>>(bufB, stats + 128, TOK);
  k_bn_apply<<<(TOK * 64 + 255) / 256, 256, 0, stream>>>(bufB, stats + 128, F(I_DG2), F(I_DBE2),
                                                         bufA, nullptr, TOK);
  // residual branch
  k_gemm<<<gB, tB, 0, stream>>>(hK, wh + O_RWT, F(I_RB), bufC, nullptr, TOK, 64, 64, 0);
  k_bn_stats<<<64, 256, 0, stream>>>(bufC, stats + 256, TOK);
  k_bn_apply<<<(TOK * 64 + 255) / 256, 256, 0, stream>>>(bufC, stats + 256, F(I_RG), F(I_RBE),
                                                         bufC, nullptr, TOK);
  // combine + recon loss
  k_combine<<<(TOK + 255) / 256, 256, 0, stream>>>(bufA, bufC, F(I_OW), F(I_OB), F(I_X), acc + 3);

  // remaining outputs
  k_dists<<<207, 64, 0, stream>>>(bufD, F(I_EMB2), out_dists);
  hipMemcpyAsync(out_centers, d_in[I_EMB2], 4096 * sizeof(float),
                 hipMemcpyDeviceToDevice, stream);
  k_final<<<1, 1, 0, stream>>>(acc, out_scal);
}
